// FinancialReasoningGNN_27522150433049
// MI455X (gfx1250) — matrix operand order repaired and checked by reading, not compiled, not run on hardware
//
#include <hip/hip_runtime.h>

typedef __attribute__((ext_vector_type(16))) _Float16 v16h;
typedef __attribute__((ext_vector_type(8)))  float    v8f;

#define HIDC 64
#define H1   192   // HID*IN : nnconv hidden width and per-edge weight width
#define W2LD 200   // padded LDS row stride (f16): 400B rows -> conflict-free banks
#define MRGH 128
#define MRGK 320
#define OUTC 32

struct U32x8 { uint4 lo, hi; };   // 32B carrier for bit_cast to v16h

__device__ __forceinline__ unsigned enc_f(float f) {
    unsigned u = __float_as_uint(f);
    return (u & 0x80000000u) ? ~u : (u | 0x80000000u);
}
__device__ __forceinline__ float dec_f(unsigned e) {
    unsigned u = (e & 0x80000000u) ? (e ^ 0x80000000u) : ~e;
    return __uint_as_float(u);
}

// ---------------- node prep: root proj, GAT left/right proj, zero accumulators ----
__global__ void k_prep_nodes(const float* __restrict__ x,
                             const float* __restrict__ nroot, const float* __restrict__ nbias,
                             const float* __restrict__ gwl, const float* __restrict__ gbl,
                             const float* __restrict__ gwr, const float* __restrict__ gbr,
                             float* __restrict__ hs, float* __restrict__ xl, float* __restrict__ xr,
                             float* __restrict__ gacc, float* __restrict__ cnt, float* __restrict__ sums,
                             unsigned* __restrict__ mxe, float* __restrict__ den, int N)
{
    int n = blockIdx.x * blockDim.x + threadIdx.x;
    if (n >= N) return;
    float x0 = x[n * 3 + 0], x1 = x[n * 3 + 1], x2 = x[n * 3 + 2];
    for (int h = 0; h < HIDC; ++h)
        hs[n * HIDC + h] = nroot[h * 3] * x0 + nroot[h * 3 + 1] * x1 + nroot[h * 3 + 2] * x2 + nbias[h];
    for (int k = 0; k < 256; ++k) {
        xl[n * 256 + k] = gwl[k * 3] * x0 + gwl[k * 3 + 1] * x1 + gwl[k * 3 + 2] * x2 + gbl[k];
        xr[n * 256 + k] = gwr[k * 3] * x0 + gwr[k * 3 + 1] * x1 + gwr[k * 3 + 2] * x2 + gbr[k];
        gacc[n * 256 + k] = 0.f;
    }
    cnt[n] = 0.f; sums[n] = 0.f;
    for (int h = 0; h < 4; ++h) { mxe[n * 4 + h] = 0u; den[n * 4 + h] = 0.f; }
}

// ---------------- fp32 -> f16 weight conversion (WMMA B operands) -----------------
__global__ void k_convert(const float* __restrict__ w2, const float* __restrict__ mw1,
                          _Float16* __restrict__ w2h, _Float16* __restrict__ w1h)
{
    int i = blockIdx.x * blockDim.x + threadIdx.x;
    if (i < H1 * H1)       w2h[i] = (_Float16)w2[i];
    if (i < MRGH * MRGK)   w1h[i] = (_Float16)mw1[i];
}

// ---------------- in-degree count + edge-attr sums (self-loop fill = mean) --------
__global__ void k_deg(const int* __restrict__ dst, const float* __restrict__ attr,
                      float* __restrict__ cnt, float* __restrict__ sums, int E)
{
    int e = blockIdx.x * blockDim.x + threadIdx.x;
    if (e >= E) return;
    int d = dst[e];
    atomicAdd(&cnt[d], 1.f);
    atomicAdd(&sums[d], attr[e]);
}

// ---------------- fused NNConv ----------------------------------------------------
// edge-MLP layer1 built straight into WMMA A-fragments; W2 (f16) staged once per
// block into LDS (74KB > 64KB WGP$, so global streaming would thrash to L2);
// 72 WMMAs per 16-edge wave tile; contract vs x[src]; atomic segment-sum into L2.
__global__ void __launch_bounds__(256)
k_nnconv(const float* __restrict__ x, const int* __restrict__ src, const int* __restrict__ dst,
         const float* __restrict__ attr, const float* __restrict__ w1, const float* __restrict__ b1,
         const _Float16* __restrict__ w2h, const float* __restrict__ b2,
         float* __restrict__ hs, int E)
{
    extern __shared__ _Float16 w2s[];     // [192][W2LD] = 76,800 B dynamic LDS

    // cooperative stage: 36,864 f16 in 16B chunks, re-laid to padded rows
    for (int i = threadIdx.x; i < (H1 * H1) / 8; i += 256) {
        int row = i / (H1 / 8);
        int col = (i % (H1 / 8)) * 8;
        *(uint4*)(w2s + row * W2LD + col) = *(const uint4*)(w2h + row * H1 + col);
    }
    __syncthreads();

    int wave = threadIdx.x >> 5;
    int lane = threadIdx.x & 31;
    int tile = blockIdx.x * 8 + wave;
    int e0 = tile * 16;
    if (e0 >= E) return;                  // wave-uniform: EXEC all-ones for WMMA
    int l16 = lane & 15;
    int hh  = lane >> 4;

    int eA = e0 + l16; if (eA >= E) eA = E - 1;
    float a = attr[eA];
    __builtin_prefetch(attr + e0 + 2048, 0, 1);

    v8f acc[12] = {};
    #pragma unroll
    for (int kc = 0; kc < 6; ++kc) {
        // A fragment: hrelu[e, j] = relu(a*w1[j] + b1[j]) in contiguous k order (k = 16*hh + s)
        v16h af;
        #pragma unroll
        for (int s = 0; s < 16; ++s) {
            int j = kc * 32 + hh * 16 + s;
            float v = fmaf(a, w1[j], b1[j]);
            af[s] = (_Float16)(v > 0.f ? v : 0.f);
        }
        #pragma unroll
        for (int nc = 0; nc < 12; ++nc) {
            // B fragment from LDS: row n = nc*16+l16 (distinct banks via W2LD pad),
            // 16 contiguous f16 at k-base kc*32 + hh*16 -> two 16B ds loads
            const _Float16* bp = w2s + (nc * 16 + l16) * W2LD + kc * 32 + hh * 16;
            U32x8 r; r.lo = *(const uint4*)bp; r.hi = *(const uint4*)(bp + 8);
            v16h bf = __builtin_bit_cast(v16h, r);
            acc[nc] = __builtin_amdgcn_wmma_f32_16x16x32_f16(
                false, af, false, bf, (short)0, acc[nc], false, false);
        }
    }
    #pragma unroll
    for (int nc = 0; nc < 12; ++nc) {
        float bb = b2[nc * 16 + l16];
        #pragma unroll
        for (int v = 0; v < 8; ++v) acc[nc][v] += bb;
    }
    // contract against x[src] over IN=3 (D layout: lane N=l16, M = v + 8*hh)
    #pragma unroll
    for (int v = 0; v < 8; ++v) {
        int e = e0 + v + 8 * hh;
        int ec = (e < E) ? e : (E - 1);
        int s = src[ec], d = dst[ec];
        float xs0 = x[s * 3], xs1 = x[s * 3 + 1], xs2 = x[s * 3 + 2];
        #pragma unroll
        for (int t = 0; t < 4; ++t) {
            float m = xs0 * acc[t][v] + xs1 * acc[t + 4][v] + xs2 * acc[t + 8][v];
            if (e < E) atomicAdd(&hs[d * HIDC + t * 16 + l16], m);
        }
    }
}

// ---------------- GATv2 pass 1: logits + segment max (monotone-uint atomicMax) ----
__global__ void k_gat_logits(const int* __restrict__ src, const int* __restrict__ dst,
                             const float* __restrict__ attr, const float* __restrict__ cnt,
                             const float* __restrict__ sums, const float* __restrict__ xl,
                             const float* __restrict__ xr, const float* __restrict__ we,
                             const float* __restrict__ att, float* __restrict__ logits,
                             unsigned* __restrict__ mxe, int E, int N)
{
    int t = blockIdx.x * blockDim.x + threadIdx.x;
    if (t >= (E + N) * 4) return;
    int ea = t >> 2, hd = t & 3;
    int s, d; float a;
    if (ea < E) { s = src[ea]; d = dst[ea]; a = attr[ea]; }
    else        { int n = ea - E; s = n; d = n; a = sums[n] / fmaxf(cnt[n], 1.f); }
    const float* xls = xl + s * 256 + hd * 64;
    const float* xrd = xr + d * 256 + hd * 64;
    const float* wep = we + hd * 64;
    const float* atp = att + hd * 64;
    float lg = 0.f;
    for (int c = 0; c < 64; ++c) {
        float m = xls[c] + xrd[c] + a * wep[c];
        m = (m > 0.f) ? m : 0.2f * m;
        lg += m * atp[c];
    }
    logits[t] = lg;
    atomicMax(&mxe[d * 4 + hd], enc_f(lg));
}

// ---------------- GATv2 pass 2: exp(logit - max), segment sum ---------------------
__global__ void k_gat_ex(const int* __restrict__ dst, float* __restrict__ logits,
                         const unsigned* __restrict__ mxe, float* __restrict__ den, int E, int N)
{
    int t = blockIdx.x * blockDim.x + threadIdx.x;
    if (t >= (E + N) * 4) return;
    int ea = t >> 2, hd = t & 3;
    int d = (ea < E) ? dst[ea] : (ea - E);
    float ex = __expf(logits[t] - dec_f(mxe[d * 4 + hd]));
    logits[t] = ex;
    atomicAdd(&den[d * 4 + hd], ex);
}

// ---------------- GATv2 pass 3: alpha-weighted aggregation ------------------------
__global__ void k_gat_aggr(const int* __restrict__ src, const int* __restrict__ dst,
                           const float* __restrict__ exv, const float* __restrict__ den,
                           const float* __restrict__ xl, float* __restrict__ gacc, int E, int N)
{
    long long t = (long long)blockIdx.x * blockDim.x + threadIdx.x;
    if (t >= (long long)(E + N) * 64) return;
    int ea = (int)(t >> 6), c = (int)(t & 63);
    int s, d;
    if (ea < E) { s = src[ea]; d = dst[ea]; } else { s = ea - E; d = s; }
    #pragma unroll
    for (int hd = 0; hd < 4; ++hd) {
        float alpha = exv[ea * 4 + hd] / (den[d * 4 + hd] + 1e-16f);
        atomicAdd(&gacc[d * 256 + hd * 64 + c], xl[s * 256 + hd * 64 + c] * alpha);
    }
}

// ---------------- merger MLP: layer-1 via WMMA (K=320,N=128), layer-2 VALU --------
__global__ void __launch_bounds__(256)
k_merge(const float* __restrict__ hs, const float* __restrict__ gacc,
        const float* __restrict__ gbias, const _Float16* __restrict__ w1h,
        const float* __restrict__ b1, const float* __restrict__ w2,
        const float* __restrict__ b2, float* __restrict__ out, int N)
{
    __shared__ _Float16 z1s[8][16][MRGH];     // 32 KB: per-wave 16x128 hidden tile (f16)
    int wave = threadIdx.x >> 5;
    int lane = threadIdx.x & 31;
    int tile = blockIdx.x * 8 + wave;
    int n0 = tile * 16;
    if (n0 >= N) return;                      // wave-uniform
    int l16 = lane & 15, hh = lane >> 4;
    int node = n0 + l16; if (node >= N) node = N - 1;

    v8f acc[8] = {};
    #pragma unroll
    for (int kc = 0; kc < 10; ++kc) {
        // A: combined = [relu(h_struct) | relu(gacc + gat_bias)] in A layout
        v16h af;
        #pragma unroll
        for (int s = 0; s < 16; ++s) {
            int k = kc * 32 + hh * 16 + s;
            float v;
            if (k < 64) v = hs[node * 64 + k];
            else { int q = k - 64; v = gacc[node * 256 + q] + gbias[q]; }
            v = v > 0.f ? v : 0.f;
            af[s] = (_Float16)v;
        }
        #pragma unroll
        for (int nc = 0; nc < 8; ++nc) {
            const v16h bf = *(const v16h*)(w1h + (nc * 16 + l16) * MRGK + kc * 32 + hh * 16);
            acc[nc] = __builtin_amdgcn_wmma_f32_16x16x32_f16(
                false, af, false, bf, (short)0, acc[nc], false, false);
        }
    }
    #pragma unroll
    for (int nc = 0; nc < 8; ++nc) {
        int n = nc * 16 + l16;
        float bb = b1[n];
        #pragma unroll
        for (int v = 0; v < 8; ++v) {
            float z = acc[nc][v] + bb;
            z1s[wave][v + 8 * hh][n] = (_Float16)(z > 0.f ? z : 0.f);
        }
    }
    asm volatile("s_wait_dscnt 0" ::: "memory");  // same-wave DS store->load ordering
    int o = lane;
    for (int m = 0; m < 16; ++m) {
        if (n0 + m >= N) break;
        float av = b2[o];
        for (int k = 0; k < MRGH; ++k)
            av = fmaf((float)z1s[wave][m][k], w2[o * MRGH + k], av);
        out[(n0 + m) * OUTC + o] = av;
    }
}

extern "C" void kernel_launch(void* const* d_in, const int* in_sizes, int n_in,
                              void* d_out, int out_size, void* d_ws, size_t ws_size,
                              hipStream_t stream)
{
    const float* x     = (const float*)d_in[0];
    const int*   ei    = (const int*)  d_in[1];
    const float* attr  = (const float*)d_in[2];
    const float* nw1   = (const float*)d_in[3];
    const float* nb1   = (const float*)d_in[4];
    const float* nw2   = (const float*)d_in[5];
    const float* nb2   = (const float*)d_in[6];
    const float* nroot = (const float*)d_in[7];
    const float* nbias = (const float*)d_in[8];
    const float* gwl   = (const float*)d_in[9];
    const float* gbl   = (const float*)d_in[10];
    const float* gwr   = (const float*)d_in[11];
    const float* gbr   = (const float*)d_in[12];
    const float* gwe   = (const float*)d_in[13];
    const float* gatt  = (const float*)d_in[14];
    const float* gbias = (const float*)d_in[15];
    const float* mw1   = (const float*)d_in[16];
    const float* mb1   = (const float*)d_in[17];
    const float* mw2   = (const float*)d_in[18];
    const float* mb2   = (const float*)d_in[19];

    int N  = in_sizes[0] / 3;
    int E  = in_sizes[2];
    int Ea = E + N;
    const int* src = ei;
    const int* dst = ei + E;

    char* p = (char*)d_ws;
    auto carve = [&](size_t bytes) { char* r = p; p += (bytes + 255) & ~(size_t)255; return r; };
    float*    hs   = (float*)   carve((size_t)N * 64 * 4);
    float*    xl   = (float*)   carve((size_t)N * 256 * 4);
    float*    xr   = (float*)   carve((size_t)N * 256 * 4);
    float*    gacc = (float*)   carve((size_t)N * 256 * 4);
    float*    cnt  = (float*)   carve((size_t)N * 4);
    float*    sums = (float*)   carve((size_t)N * 4);
    unsigned* mxe  = (unsigned*)carve((size_t)N * 4 * 4);
    float*    den  = (float*)   carve((size_t)N * 4 * 4);
    float*    exv  = (float*)   carve((size_t)Ea * 4 * 4);
    _Float16* w2h  = (_Float16*)carve((size_t)H1 * H1 * 2);
    _Float16* w1h  = (_Float16*)carve((size_t)MRGH * MRGK * 2);
    (void)ws_size; (void)n_in; (void)out_size;

    k_prep_nodes<<<(N + 255) / 256, 256, 0, stream>>>(x, nroot, nbias, gwl, gbl, gwr, gbr,
                                                      hs, xl, xr, gacc, cnt, sums, mxe, den, N);
    int ncv = MRGH * MRGK;   // >= H1*H1
    k_convert<<<(ncv + 255) / 256, 256, 0, stream>>>(nw2, mw1, w2h, w1h);
    k_deg<<<(E + 255) / 256, 256, 0, stream>>>(dst, attr, cnt, sums, E);

    int etiles = (E + 15) / 16;
    int ldsBytes = H1 * W2LD * (int)sizeof(_Float16);   // 76,800 B dynamic LDS
    hipFuncSetAttribute((const void*)k_nnconv, hipFuncAttributeMaxDynamicSharedMemorySize, ldsBytes);
    k_nnconv<<<(etiles + 7) / 8, 256, ldsBytes, stream>>>(x, src, dst, attr, nw1, nb1, w2h, nb2, hs, E);

    int tl = Ea * 4;
    k_gat_logits<<<(tl + 255) / 256, 256, 0, stream>>>(src, dst, attr, cnt, sums, xl, xr,
                                                       gwe, gatt, exv, mxe, E, N);
    k_gat_ex<<<(tl + 255) / 256, 256, 0, stream>>>(dst, exv, mxe, den, E, N);

    long long ta = (long long)Ea * 64;
    k_gat_aggr<<<(unsigned)((ta + 255) / 256), 256, 0, stream>>>(src, dst, exv, den, xl, gacc, E, N);

    int ntiles = (N + 15) / 16;
    k_merge<<<(ntiles + 7) / 8, 256, 0, stream>>>(hs, gacc, gbias, w1h, mb1, mw2, mb2,
                                                  (float*)d_out, N);
}
